// PersimmonAttention_4337916969006
// MI455X (gfx1250) — matrix operand order repaired and checked by reading, not compiled, not run on hardware
//
#include <hip/hip_runtime.h>
#include <hip/hip_bf16.h>

#define SEQ 2048
#define HID 4096
#define NHD 64
#define HD  64

typedef __attribute__((ext_vector_type(16))) __bf16         v16bf;
typedef __attribute__((ext_vector_type(8)))  float          v8f;
typedef __attribute__((ext_vector_type(8)))  unsigned short ushort8;
typedef int v4i_vs __attribute__((vector_size(16)));   // gcc-vector int4 (builtin param type)

#if defined(__has_builtin)
#if __has_builtin(__builtin_amdgcn_global_load_async_to_lds_b128) && \
    __has_builtin(__builtin_amdgcn_s_wait_asynccnt)
#define HAS_ASYNC_LDS 1
#endif
#endif
#ifndef HAS_ASYNC_LDS
#define HAS_ASYNC_LDS 0
#endif

union Frag {
    ushort8 h[2];   // two 16-byte halves
    v16bf   v;      // WMMA operand view
};

__device__ __forceinline__ unsigned short f2bf(float f) {
    unsigned int u = __float_as_uint(f);
    unsigned int r = u + 0x7fffu + ((u >> 16) & 1u);   // round-to-nearest-even
    return (unsigned short)(r >> 16);
}

__device__ __forceinline__ v8f wmma_bf16(const Frag& a, const Frag& b, v8f c) {
    return __builtin_amdgcn_wmma_f32_16x16x32_bf16(
        false, a.v, false, b.v, (short)0, c, false, false);
}

// ---------------------------------------------------------------------------
// Elementwise f32 -> bf16 cast (hidden states).
// ---------------------------------------------------------------------------
__global__ __launch_bounds__(256) void cast_kernel(
    const float* __restrict__ src, unsigned short* __restrict__ dst, int n)
{
    int i = (blockIdx.x * 256 + threadIdx.x) * 4;
    if (i + 3 < n) {
#pragma unroll
        for (int j = 0; j < 4; ++j) dst[i + j] = f2bf(src[i + j]);
    }
}

// ---------------------------------------------------------------------------
// Cast + transpose: W[K][N] f32 -> Wt[N][K] bf16.  32x32 tiles through LDS.
// One-time pass so every GEMM tile-stage becomes a contiguous bf16 copy and
// the bf16 weights (96/32 MB) live in the 192 MB L2 across M-block re-reads.
// ---------------------------------------------------------------------------
__global__ __launch_bounds__(256) void transpose_cast_kernel(
    const float* __restrict__ W, unsigned short* __restrict__ Wt, int K, int N)
{
    __shared__ unsigned short tile[32][33];
    const int k0 = blockIdx.y * 32, n0 = blockIdx.x * 32;
    const int r = threadIdx.x >> 3, c = (threadIdx.x & 7) * 4;
    const float* src = W + (size_t)(k0 + r) * N + n0 + c;
#pragma unroll
    for (int i = 0; i < 4; ++i) tile[r][c + i] = f2bf(src[i]);
    __syncthreads();
    unsigned short* dst = Wt + (size_t)(n0 + r) * K + k0 + c;
#pragma unroll
    for (int i = 0; i < 4; ++i) dst[i] = tile[c + i][r];
}

// ---------------------------------------------------------------------------
// All-bf16 GEMM: C[M,N] f32 = A[M,K] @ Bt[N,K]^T + bias.
// 128x128 tile, Ktile=32, 8 waves x (64x32). Double-buffered LDS staged with
// GLOBAL_LOAD_ASYNC_TO_LDS_B128 (ASYNCcnt) when available; register-staged
// single-buffer fallback otherwise. Fragments match ISA bf16 WMMA layouts:
//   A lane: k = 8*half + e (e<8), +16 for e>=8  -> two 16B chunks
//   B lane: k = 8*half + e (e<8), +16 for e>=8  -> two 16B chunks
// ---------------------------------------------------------------------------
__global__ __launch_bounds__(256) void gemm_bf16_kernel(
    const unsigned short* __restrict__ A,   // [M][K] bf16
    const unsigned short* __restrict__ Bt,  // [N][K] bf16 (pre-transposed)
    const float* __restrict__ bias, float* __restrict__ C,
    int M, int N, int K)
{
    __shared__ unsigned short As[2][128 * 32];
    __shared__ unsigned short Bs[2][128 * 32];

    const int tid    = threadIdx.x;
    const int w      = tid >> 5, lane = tid & 31;
    const int lane16 = lane & 15, hl = lane >> 4;
    const int wm     = w & 1, wn = w >> 1;
    const int m0     = blockIdx.y * 128, n0 = blockIdx.x * 128;
    const int row    = tid >> 1, cg = (tid & 1) * 16;

    const unsigned short* gA = A  + (size_t)(m0 + row) * K + cg;
    const unsigned short* gB = Bt + (size_t)(n0 + row) * K + cg;
    const int nk = K >> 5;

    v8f acc[4][2] = {};

    auto compute = [&](const unsigned short* as, const unsigned short* bs) {
        Frag af[4], bf[2];
#pragma unroll
        for (int mt = 0; mt < 4; ++mt) {
            const unsigned short* p = as + (wm * 64 + mt * 16 + lane16) * 32 + 8 * hl;
            af[mt].h[0] = *(const ushort8*)(p);
            af[mt].h[1] = *(const ushort8*)(p + 16);
        }
#pragma unroll
        for (int nt = 0; nt < 2; ++nt) {
            const unsigned short* p = bs + (wn * 32 + nt * 16 + lane16) * 32 + 8 * hl;
            bf[nt].h[0] = *(const ushort8*)(p);
            bf[nt].h[1] = *(const ushort8*)(p + 16);
        }
#pragma unroll
        for (int mt = 0; mt < 4; ++mt)
#pragma unroll
            for (int nt = 0; nt < 2; ++nt)
                acc[mt][nt] = wmma_bf16(af[mt], bf[nt], acc[mt][nt]);
    };

#if HAS_ASYNC_LDS
    auto stage = [&](int kt, int b) {
        const unsigned short* sa = gA + kt * 32;
        const unsigned short* sb = gB + kt * 32;
        unsigned short* da = &As[b][row * 32 + cg];
        unsigned short* db = &Bs[b][row * 32 + cg];
        __builtin_amdgcn_global_load_async_to_lds_b128(
            (__attribute__((address_space(1))) v4i_vs*)(void*)(sa),
            (__attribute__((address_space(3))) v4i_vs*)(void*)(da), 0, 0);
        __builtin_amdgcn_global_load_async_to_lds_b128(
            (__attribute__((address_space(1))) v4i_vs*)(void*)(sa + 8),
            (__attribute__((address_space(3))) v4i_vs*)(void*)(da + 8), 0, 0);
        __builtin_amdgcn_global_load_async_to_lds_b128(
            (__attribute__((address_space(1))) v4i_vs*)(void*)(sb),
            (__attribute__((address_space(3))) v4i_vs*)(void*)(db), 0, 0);
        __builtin_amdgcn_global_load_async_to_lds_b128(
            (__attribute__((address_space(1))) v4i_vs*)(void*)(sb + 8),
            (__attribute__((address_space(3))) v4i_vs*)(void*)(db + 8), 0, 0);
    };
    stage(0, 0);
    if (nk > 1) stage(1, 1);
    for (int kt = 0; kt < nk; ++kt) {
        if (kt + 1 < nk) __builtin_amdgcn_s_wait_asynccnt(4);  // tile kt landed
        else             __builtin_amdgcn_s_wait_asynccnt(0);
        __syncthreads();
        compute(As[kt & 1], Bs[kt & 1]);
        __syncthreads();                      // everyone done reading buf kt&1
        if (kt + 2 < nk) stage(kt + 2, kt & 1);
    }
#else
    ushort8 ra0, ra1, rb0, rb1;
    auto ldregs = [&](int kt) {
        const unsigned short* sa = gA + kt * 32;
        const unsigned short* sb = gB + kt * 32;
        ra0 = *(const ushort8*)(sa); ra1 = *(const ushort8*)(sa + 8);
        rb0 = *(const ushort8*)(sb); rb1 = *(const ushort8*)(sb + 8);
    };
    ldregs(0);
    for (int kt = 0; kt < nk; ++kt) {
        __syncthreads();
        *(ushort8*)&As[0][row * 32 + cg]     = ra0;
        *(ushort8*)&As[0][row * 32 + cg + 8] = ra1;
        *(ushort8*)&Bs[0][row * 32 + cg]     = rb0;
        *(ushort8*)&Bs[0][row * 32 + cg + 8] = rb1;
        __syncthreads();
        if (kt + 1 < nk) ldregs(kt + 1);      // prefetch overlaps compute
        compute(As[0], Bs[0]);
    }
#endif

    // epilogue: C layout VGPR r -> (m = r + 8*hl, n = lane16)
#pragma unroll
    for (int mt = 0; mt < 4; ++mt)
#pragma unroll
        for (int nt = 0; nt < 2; ++nt) {
            const int col = n0 + wn * 32 + nt * 16 + lane16;
            const float bv = bias[col];
#pragma unroll
            for (int r = 0; r < 8; ++r) {
                const int rr = m0 + wm * 64 + mt * 16 + r + 8 * hl;
                C[(size_t)rr * N + col] = acc[mt][nt][r] + bv;
            }
        }
}

// ---------------------------------------------------------------------------
// Per-(s,head) LayerNorm(q,k) + RoPE + scale. One wave32 per (s,head).
// Writes Q,K bf16 [NH][S][HD] and V transposed bf16 [NH][HD][S] so the
// attention PV B-fragments are contiguous global loads (no LDS staging).
// ---------------------------------------------------------------------------
__global__ __launch_bounds__(128) void qkv_post_kernel(
    const float* __restrict__ fused, const int* __restrict__ pos_ids,
    const float* __restrict__ qw, const float* __restrict__ qb,
    const float* __restrict__ kw, const float* __restrict__ kb,
    unsigned short* __restrict__ Q, unsigned short* __restrict__ Kd,
    unsigned short* __restrict__ Vt)
{
    const int wid  = blockIdx.x * 4 + (threadIdx.x >> 5);
    const int lane = threadIdx.x & 31;
    const int s = wid / NHD, h = wid % NHD;

    const float* base = fused + (size_t)s * (3 * HID) + h * (3 * HD);
    float q0 = base[lane],       q1 = base[lane + 32];
    float k0 = base[64 + lane],  k1 = base[64 + lane + 32];
    float v0 = base[128 + lane], v1 = base[128 + lane + 32];

    auto ln = [&](float& a, float& b, const float* w_, const float* b_) {
        float sum = a + b;
#pragma unroll
        for (int off = 16; off; off >>= 1) sum += __shfl_xor(sum, off, 32);
        const float mu = sum * (1.f / 64.f);
        float da = a - mu, db = b - mu;
        float vs = da * da + db * db;
#pragma unroll
        for (int off = 16; off; off >>= 1) vs += __shfl_xor(vs, off, 32);
        const float inv = rsqrtf(vs * (1.f / 64.f) + 1e-5f);
        a = da * inv * w_[lane]      + b_[lane];
        b = db * inv * w_[lane + 32] + b_[lane + 32];
    };
    ln(q0, q1, qw, qb);
    ln(k0, k1, kw, kb);

    // RoPE on d = lane (0..31): partner d^16, emb index = d & 15
    const float pos = (float)pos_ids[s];
    const float inv_freq = __expf(-((float)(2 * (lane & 15)) / 32.f) * __logf(25000.f));
    const float ang = pos * inv_freq;
    const float c = __cosf(ang), sn = __sinf(ang);
    const float qp = __shfl_xor(q0, 16, 32);
    const float kp = __shfl_xor(k0, 16, 32);
    const float sgn = (lane < 16) ? -1.f : 1.f;
    q0 = q0 * c + sgn * qp * sn;
    k0 = k0 * c + sgn * kp * sn;

    const float sc = 0.125f;                      // 1/sqrt(HD) folded into Q
    q0 *= sc; q1 *= sc;

    const size_t o = ((size_t)h * SEQ + s) * HD;
    Q[o + lane] = f2bf(q0);  Q[o + lane + 32] = f2bf(q1);
    Kd[o + lane] = f2bf(k0); Kd[o + lane + 32] = f2bf(k1);
    Vt[((size_t)h * HD + lane)      * SEQ + s] = f2bf(v0);
    Vt[((size_t)h * HD + lane + 32) * SEQ + s] = f2bf(v1);
}

// ---------------------------------------------------------------------------
// Flash attention per (head, 128 query rows). 8 independent waves x 16 rows.
// K^T and V^T fragments load straight from global (two 16B chunks per lane);
// P goes C-layout -> per-wave LDS scratch -> A-layout. No block barriers.
// ---------------------------------------------------------------------------
__global__ __launch_bounds__(256) void attn_kernel(
    const unsigned short* __restrict__ Q, const unsigned short* __restrict__ Kd,
    const unsigned short* __restrict__ Vt, unsigned short* __restrict__ Oattn)
{
    __shared__ unsigned short Ps[8][16 * 32];     // per-wave P scratch [m][k]

    const int tid = threadIdx.x;
    const int w = tid >> 5, lane = tid & 31;
    const int lane16 = lane & 15, hl = lane >> 4;
    const int h = blockIdx.y;
    const int q0row = blockIdx.x * 128;
    const int qrow = q0row + w * 16;

    Frag qf[2];
#pragma unroll
    for (int kk = 0; kk < 2; ++kk) {
        const unsigned short* p =
            Q + ((size_t)h * SEQ + qrow + lane16) * HD + kk * 32 + 8 * hl;
        qf[kk].h[0] = *(const ushort8*)(p);
        qf[kk].h[1] = *(const ushort8*)(p + 16);
    }

    v8f o[4] = {};
    float mrow[8], lrow[8];
#pragma unroll
    for (int r = 0; r < 8; ++r) { mrow[r] = -3.0e38f; lrow[r] = 0.f; }

    const int nkb = (q0row + 128) / 32;
    for (int kb = 0; kb < nkb; ++kb) {
        const int kbase = kb * 32;

        // ---- S = Q K^T : two 16x16 f32 tiles over keys [kbase, kbase+32) ----
        v8f s0 = {}, s1 = {};
#pragma unroll
        for (int kk = 0; kk < 2; ++kk) {
            Frag b0, b1;
            const unsigned short* p0 =
                Kd + ((size_t)h * SEQ + kbase + lane16) * HD + kk * 32 + 8 * hl;
            b0.h[0] = *(const ushort8*)(p0);
            b0.h[1] = *(const ushort8*)(p0 + 16);
            const unsigned short* p1 = p0 + (size_t)16 * HD;
            b1.h[0] = *(const ushort8*)(p1);
            b1.h[1] = *(const ushort8*)(p1 + 16);
            s0 = wmma_bf16(qf[kk], b0, s0);
            s1 = wmma_bf16(qf[kk], b1, s1);
        }

        // ---- causal mask + online softmax (row = r + 8*hl over 16 lanes) ----
        float pv0[8], pv1[8], rscale[8];
#pragma unroll
        for (int r = 0; r < 8; ++r) {
            const int qi  = qrow + r + 8 * hl;
            const int k0i = kbase + lane16, k1i = k0i + 16;
            const bool ok0 = (k0i <= qi), ok1 = (k1i <= qi);
            float v0s = ok0 ? s0[r] : -3.0e38f;
            float v1s = ok1 ? s1[r] : -3.0e38f;
            float rm = fmaxf(v0s, v1s);
#pragma unroll
            for (int off = 1; off < 16; off <<= 1) rm = fmaxf(rm, __shfl_xor(rm, off, 32));
            const float mnew = fmaxf(mrow[r], rm);
            const float p0e = ok0 ? __expf(s0[r] - mnew) : 0.f;
            const float p1e = ok1 ? __expf(s1[r] - mnew) : 0.f;
            const float scale = __expf(mrow[r] - mnew);
            float ls = p0e + p1e;
#pragma unroll
            for (int off = 1; off < 16; off <<= 1) ls += __shfl_xor(ls, off, 32);
            lrow[r] = lrow[r] * scale + ls;
            mrow[r] = mnew;
            rscale[r] = scale;
            pv0[r] = p0e; pv1[r] = p1e;
        }
#pragma unroll
        for (int dt = 0; dt < 4; ++dt)
#pragma unroll
            for (int r = 0; r < 8; ++r) o[dt][r] *= rscale[r];

        // ---- P: C-layout -> per-wave LDS -> A-layout fragment ----
        unsigned short* pw = &Ps[w][0];
#pragma unroll
        for (int r = 0; r < 8; ++r) {
            const int m = r + 8 * hl;
            pw[m * 32 + lane16]      = f2bf(pv0[r]);
            pw[m * 32 + lane16 + 16] = f2bf(pv1[r]);
        }
        Frag pf;
        const unsigned short* pr = &Ps[w][lane16 * 32 + 8 * hl];
        pf.h[0] = *(const ushort8*)(pr);
        pf.h[1] = *(const ushort8*)(pr + 16);

        // ---- O += P @ V  (V^T fragments straight from global) ----
#pragma unroll
        for (int dt = 0; dt < 4; ++dt) {
            Frag vf;
            const unsigned short* vp =
                Vt + ((size_t)h * HD + dt * 16 + lane16) * SEQ + kbase + 8 * hl;
            vf.h[0] = *(const ushort8*)(vp);
            vf.h[1] = *(const ushort8*)(vp + 16);
            o[dt] = wmma_bf16(pf, vf, o[dt]);
        }
    }

#pragma unroll
    for (int r = 0; r < 8; ++r) {
        const float inv = 1.f / lrow[r];
        const int m = qrow + r + 8 * hl;
#pragma unroll
        for (int dt = 0; dt < 4; ++dt) {
            const int d = h * HD + dt * 16 + lane16;
            Oattn[(size_t)m * HID + d] = f2bf(o[dt][r] * inv);
        }
    }
}

// ---------------------------------------------------------------------------
extern "C" void kernel_launch(void* const* d_in, const int* in_sizes, int n_in,
                              void* d_out, int out_size, void* d_ws, size_t ws_size,
                              hipStream_t stream)
{
    const float* hidden = (const float*)d_in[0];
    // d_in[1] = attention_mask (pure causal; reproduced inline)
    const int*   pos    = (const int*)d_in[2];
    const float* Wqkv   = (const float*)d_in[3];
    const float* bqkv   = (const float*)d_in[4];
    const float* Wd     = (const float*)d_in[5];
    const float* bd     = (const float*)d_in[6];
    const float* qlw    = (const float*)d_in[7];
    const float* qlb    = (const float*)d_in[8];
    const float* klw    = (const float*)d_in[9];
    const float* klb    = (const float*)d_in[10];
    float* out = (float*)d_out;

    const size_t MB = 1024 * 1024;
    char* ws = (char*)d_ws;
    float*          fused  = (float*)ws;                                  //  96 MiB
    unsigned short* hidB   = (unsigned short*)(ws + 96  * MB);            // +16 MiB
    unsigned short* WtQKV  = (unsigned short*)(ws + 112 * MB);            // +96 MiB
    unsigned short* WtD    = (unsigned short*)(ws + 208 * MB);            // +32 MiB
    unsigned short* Qb     = (unsigned short*)(ws + 240 * MB);            // +16 MiB
    unsigned short* Kb     = (unsigned short*)(ws + 256 * MB);            // +16 MiB
    unsigned short* Vtb    = (unsigned short*)(ws + 272 * MB);            // +16 MiB
    unsigned short* attnb  = (unsigned short*)(ws + 288 * MB);            // +16 MiB

    // 0) one-time precision/layout conversion
    cast_kernel<<<(SEQ * HID) / 1024, 256, 0, stream>>>(hidden, hidB, SEQ * HID);
    transpose_cast_kernel<<<dim3((3 * HID) / 32, HID / 32), 256, 0, stream>>>(
        Wqkv, WtQKV, HID, 3 * HID);
    transpose_cast_kernel<<<dim3(HID / 32, HID / 32), 256, 0, stream>>>(
        Wd, WtD, HID, HID);

    // 1) fused QKV GEMM (bf16 WMMA, f32 accum/out)
    gemm_bf16_kernel<<<dim3((3 * HID) / 128, SEQ / 128), 256, 0, stream>>>(
        hidB, WtQKV, bqkv, fused, SEQ, 3 * HID, HID);

    // 2) LayerNorm + RoPE + scale -> Q,K [NH][S][HD], V^T [NH][HD][S]
    qkv_post_kernel<<<(SEQ * NHD) / 4, 128, 0, stream>>>(
        fused, pos, qlw, qlb, klw, klb, Qb, Kb, Vtb);

    // 3) flash attention -> attn bf16 [S][HID]
    attn_kernel<<<dim3(SEQ / 128, NHD), 256, 0, stream>>>(Qb, Kb, Vtb, attnb);

    // 4) output projection
    gemm_bf16_kernel<<<dim3(HID / 128, SEQ / 128), 256, 0, stream>>>(
        attnb, WtD, bd, out, SEQ, HID, HID);
}
